// GraphAttentionLayer_7017976561798
// MI455X (gfx1250) — matrix operand order repaired and checked by reading, not compiled, not run on hardware
//
#include <hip/hip_runtime.h>
#include <hip/hip_bf16.h>

typedef __attribute__((ext_vector_type(16))) _Float16 v16h;
typedef __attribute__((ext_vector_type(8)))  _Float16 v8h;
typedef __attribute__((ext_vector_type(8)))  float    v8f;

#define GAT_B    8
#define GAT_N    2048
#define GAT_FIN  128
#define GAT_FOUT 64
#define GAT_ALPHA 0.2f
#define GAT_NEGBIG (-9.0e15f)
#define GAT_LOG2E 1.4426950408889634f

static __device__ __forceinline__ v8f wmma_f16(v16h a, v16h b, v8f c) {
  // 8 args: (neg_a, A, neg_b, B, c_mod, C, reuse_a, reuse_b)
  return __builtin_amdgcn_wmma_f32_16x16x32_f16(false, a, false, b, (short)0, c,
                                                false, false);
}

// ---------------------------------------------------------------------------
// Kernel 1: Wh = h @ W  (one wave per 16x16 tile; WMMA f16, f32 accum).
// Writes Wh (f32, [b*n][64]) and WhT (f16, [b][c][n]) for phase-2 B operands.
// ---------------------------------------------------------------------------
__global__ __launch_bounds__(32) void gat_wh_kernel(
    const float* __restrict__ h, const float* __restrict__ W,
    float* __restrict__ Wh, _Float16* __restrict__ WhT) {
  const int tm   = blockIdx.x;          // 0 .. B*N/16-1
  const int tn   = blockIdx.y;          // 0 .. 3
  const int lane = threadIdx.x;
  const int mrow = lane & 15;
  const int hi   = lane >> 4;
  const int row  = tm * 16 + mrow;      // global row in [0, B*N)
  const int col  = tn * 16 + mrow;      // output column (B/D lane column)
  const float* hrow = h + (size_t)row * GAT_FIN;

  v8f acc = {};
  for (int k0 = 0; k0 < GAT_FIN; k0 += 32) {
    // A operand (16-bit 16x32): lane<16 holds K {0..7,16..23}, lane>=16 {8..15,24..31}
    const int ka = k0 + hi * 8;
    float4 x0 = *(const float4*)(hrow + ka);
    float4 x1 = *(const float4*)(hrow + ka + 4);
    float4 x2 = *(const float4*)(hrow + ka + 16);
    float4 x3 = *(const float4*)(hrow + ka + 20);
    float af[16];
    *(float4*)(af + 0)  = x0;
    *(float4*)(af + 4)  = x1;
    *(float4*)(af + 8)  = x2;
    *(float4*)(af + 12) = x3;
    v16h A;
#pragma unroll
    for (int e = 0; e < 16; ++e) A[e] = (_Float16)af[e];

    // B operand (32x16): N=lane&15; lane<16 K {0..7,16..23}, lane>=16 {8..15,24..31}
    const int kb = k0 + hi * 8;
    v16h Bv;
#pragma unroll
    for (int e = 0; e < 16; ++e)
      Bv[e] = (_Float16)W[(size_t)(kb + e + ((e < 8) ? 0 : 8)) * GAT_FOUT + col];

    acc = wmma_f16(A, Bv, acc);
  }

  // D layout: lane column = lane&15, VGPR r -> row r + 8*hi.
  // A 16-row tile never crosses a batch boundary (2048 % 16 == 0).
  const int b  = (tm * 16) >> 11;
  const int n0 = (tm * 16) & (GAT_N - 1);
#pragma unroll
  for (int r = 0; r < 8; ++r)
    Wh[(size_t)(tm * 16 + r + 8 * hi) * GAT_FOUT + col] = acc[r];

  // Transposed f16 copy: 8 consecutive elements per lane -> one 16B store.
  v8h hv;
#pragma unroll
  for (int r = 0; r < 8; ++r) hv[r] = (_Float16)acc[r];
  *(v8h*)(WhT + ((size_t)(b * GAT_FOUT + col)) * GAT_N + n0 + 8 * hi) = hv;
}

// ---------------------------------------------------------------------------
// Kernel 2: f = Wh . a1, g = Wh . a2 (per row), pre-scaled by log2(e) so the
// attention kernel can use raw v_exp_f32 (exp2) throughout.
// lrelu(log2e * x) == log2e * lrelu(x), so scaling commutes with the
// activation and exp2(scaled - scaled_max) == exp(x - max) exactly.
// ---------------------------------------------------------------------------
__global__ __launch_bounds__(256) void gat_fg_kernel(
    const float* __restrict__ Wh, const float* __restrict__ a,
    float* __restrict__ f, float* __restrict__ g) {
  const int idx = blockIdx.x * blockDim.x + threadIdx.x;  // 0 .. B*N-1
  const float* w = Wh + (size_t)idx * GAT_FOUT;
  float sf = 0.f, sg = 0.f;
#pragma unroll
  for (int c = 0; c < GAT_FOUT; ++c) {
    const float v = w[c];
    sf += v * a[c];
    sg += v * a[GAT_FOUT + c];
  }
  f[idx] = sf * GAT_LOG2E;
  g[idx] = sg * GAT_LOG2E;
}

// ---------------------------------------------------------------------------
// Kernel 3: fused masked-softmax attention + (attention @ Wh).
// 1 block = (b, 16-row i-tile). 4 waves split j-range, flash-style online
// softmax in the log2 domain, WMMA f16 accumulation, LDS merge of partials.
// ---------------------------------------------------------------------------
__global__ __launch_bounds__(128) void gat_attn_kernel(
    const float* __restrict__ adj, const _Float16* __restrict__ WhT,
    const float* __restrict__ f, const float* __restrict__ g,
    float* __restrict__ out) {
  const int blk  = blockIdx.x;             // 0 .. B*(N/16)-1
  const int b    = blk >> 7;               // 128 i-tiles per batch
  const int i0   = (blk & 127) * 16;
  const int tid  = threadIdx.x;
  const int wave = tid >> 5;
  const int lane = tid & 31;
  const int mrow = lane & 15;
  const int hi   = lane >> 4;

  __shared__ float s_g[GAT_N];             // 8 KB: whole g row for this batch
  __shared__ float s_acc[4][4][16][16];    // 16 KB: per-wave partial tiles
  __shared__ float s_m[4][16];
  __shared__ float s_l[4][16];

  for (int j = tid; j < GAT_N; j += 128) s_g[j] = g[b * GAT_N + j];
  __syncthreads();

  const float fi = f[b * GAT_N + i0 + mrow];             // row logit (A rows)
  const float* adjrow = adj + (size_t)(i0 + mrow) * GAT_N;
  const _Float16* whtb = WhT + (size_t)b * GAT_FOUT * GAT_N;

  v8f acc0 = {}, acc1 = {}, acc2 = {}, acc3 = {};
  float mrun = -INFINITY, lrun = 0.f;

  const int jbeg = wave * (GAT_N / 4);
  for (int jb = jbeg; jb < jbeg + GAT_N / 4; jb += 32) {
    // --- scores tile (16x32) built directly in WMMA A-operand layout ---
    const int kb = jb + hi * 16;           // A: lane<16 K=e, lane>=16 K=16+e
    float4 a0 = *(const float4*)(adjrow + kb);
    float4 a1v = *(const float4*)(adjrow + kb + 4);
    float4 a2v = *(const float4*)(adjrow + kb + 8);
    float4 a3v = *(const float4*)(adjrow + kb + 12);
    __builtin_prefetch(adjrow + kb + 32, 0, 1);          // next block -> L2/L0
    float am[16];
    *(float4*)(am + 0)  = a0;
    *(float4*)(am + 4)  = a1v;
    *(float4*)(am + 8)  = a2v;
    *(float4*)(am + 12) = a3v;

    float ek[16];
#pragma unroll
    for (int e = 0; e < 16; ++e) {
      const int j = kb + e;
      float ev = fi + s_g[j];                            // log2-domain logit
      ev = (ev > 0.f) ? ev : GAT_ALPHA * ev;             // leaky_relu
      ek[e] = (am[e] > 0.f) ? ev : GAT_NEGBIG;           // adjacency mask
    }

    // --- online softmax, log2 domain (rows replicated in lanes L, L+16) ---
    float rm = ek[0];
#pragma unroll
    for (int e = 1; e < 16; ++e) rm = fmaxf(rm, ek[e]);
    rm = fmaxf(rm, __shfl_xor(rm, 16));
    const float mnew  = fmaxf(mrun, rm);
    const float alpha = __builtin_amdgcn_exp2f(mrun - mnew);

    float rs = 0.f;
    v16h P;
#pragma unroll
    for (int e = 0; e < 16; ++e) {
      const float p = __builtin_amdgcn_exp2f(ek[e] - mnew);
      rs += p;
      P[e] = (_Float16)p;
    }
    rs += __shfl_xor(rs, 16);
    lrun = lrun * alpha + rs;
    mrun = mnew;

    // rescale accumulators (D rows r+8*hi need row-alpha via shuffle)
    float sc[8];
#pragma unroll
    for (int r = 0; r < 8; ++r) sc[r] = __shfl(alpha, r + 8 * hi);
#pragma unroll
    for (int r = 0; r < 8; ++r) {
      acc0[r] *= sc[r]; acc1[r] *= sc[r]; acc2[r] *= sc[r]; acc3[r] *= sc[r];
    }

    // --- P(16x32) @ Wh(32x64): 4 WMMAs, B from transposed f16 Wh ---
    const int kcol = jb + hi * 16;         // B: lane<16 K=e, lane>=16 K=16+e
    const v16h B0 = *(const v16h*)(whtb + (size_t)(0  + mrow) * GAT_N + kcol);
    const v16h B1 = *(const v16h*)(whtb + (size_t)(16 + mrow) * GAT_N + kcol);
    const v16h B2 = *(const v16h*)(whtb + (size_t)(32 + mrow) * GAT_N + kcol);
    const v16h B3 = *(const v16h*)(whtb + (size_t)(48 + mrow) * GAT_N + kcol);
    acc0 = wmma_f16(P, B0, acc0);
    acc1 = wmma_f16(P, B1, acc1);
    acc2 = wmma_f16(P, B2, acc2);
    acc3 = wmma_f16(P, B3, acc3);
  }

  // --- stash per-wave partials ---
#pragma unroll
  for (int r = 0; r < 8; ++r) {
    const int m = r + 8 * hi;
    s_acc[wave][0][m][mrow] = acc0[r];
    s_acc[wave][1][m][mrow] = acc1[r];
    s_acc[wave][2][m][mrow] = acc2[r];
    s_acc[wave][3][m][mrow] = acc3[r];
  }
  if (lane < 16) { s_m[wave][lane] = mrun; s_l[wave][lane] = lrun; }
  __syncthreads();

  // --- merge 4 partials (log2 domain); wave t writes column-tile t ---
  const int t = wave;
#pragma unroll
  for (int r = 0; r < 8; ++r) {
    const int m = r + 8 * hi;
    float M = s_m[0][m];
#pragma unroll
    for (int w = 1; w < 4; ++w) M = fmaxf(M, s_m[w][m]);
    float L = 0.f, val = 0.f;
#pragma unroll
    for (int w = 0; w < 4; ++w) {
      const float fac = __builtin_amdgcn_exp2f(s_m[w][m] - M);
      L   += s_l[w][m] * fac;
      val += s_acc[w][t][m][mrow] * fac;
    }
    out[((size_t)(b * GAT_N + i0 + m)) * GAT_FOUT + t * 16 + mrow] = val / L;
  }
}

// ---------------------------------------------------------------------------
extern "C" void kernel_launch(void* const* d_in, const int* in_sizes, int n_in,
                              void* d_out, int out_size, void* d_ws, size_t ws_size,
                              hipStream_t stream) {
  const float* h   = (const float*)d_in[0];   // (8, 2048, 128)
  const float* adj = (const float*)d_in[1];   // (2048, 2048)
  const float* W   = (const float*)d_in[2];   // (128, 64)
  const float* a   = (const float*)d_in[3];   // (128, 1)
  float* out = (float*)d_out;                 // (8, 2048, 64)

  char* ws = (char*)d_ws;
  float*    Wh   = (float*)ws;                                  // 4 MB
  _Float16* WhT  = (_Float16*)(ws + (size_t)4 * 1024 * 1024);   // 2 MB
  float*    fbuf = (float*)(ws + (size_t)6 * 1024 * 1024);      // 64 KB
  float*    gbuf = fbuf + GAT_B * GAT_N;                        // 64 KB

  dim3 g1(GAT_B * GAT_N / 16, GAT_FOUT / 16);
  gat_wh_kernel<<<g1, 32, 0, stream>>>(h, W, Wh, WhT);
  gat_fg_kernel<<<(GAT_B * GAT_N) / 256, 256, 0, stream>>>(Wh, a, fbuf, gbuf);
  gat_attn_kernel<<<GAT_B * (GAT_N / 16), 128, 0, stream>>>(adj, WhT, fbuf, gbuf, out);
}
